// PredictorExpGraphConv_61529701482519
// MI455X (gfx1250) — compile-verified, matrix operand order checked by reading, not run on hardware
//
#include <hip/hip_runtime.h>
#include <hip/hip_bf16.h>

// ---------------------------------------------------------------------------
// Types for CDNA5 WMMA (wave32, 16x16x32 bf16 -> f32 accumulate)
// ---------------------------------------------------------------------------
typedef __bf16 bf16_t;
typedef bf16_t v16bf __attribute__((ext_vector_type(16)));
typedef bf16_t v8bf  __attribute__((ext_vector_type(8)));
typedef float  v8f   __attribute__((ext_vector_type(8)));

#define NODES_DIV 256  // IN feature dim

// ---------------------------------------------------------------------------
// Weight convert: W[kin][kout] f32  ->  Wt[kout][kin] bf16 (transposed)
// ---------------------------------------------------------------------------
__global__ void wcvt_kernel(const float* __restrict__ W, bf16_t* __restrict__ Wt,
                            int kin, int kout) {
    int i = blockIdx.x * blockDim.x + threadIdx.x;
    if (i >= kin * kout) return;
    int k = i / kout;
    int n = i - k * kout;
    Wt[(size_t)n * kin + k] = (bf16_t)W[i];
}

// ---------------------------------------------------------------------------
// Degree count: cnt[dst[e]] += 1
// ---------------------------------------------------------------------------
__global__ void count_kernel(const int* __restrict__ dst, float* __restrict__ cnt, int E) {
    int e = blockIdx.x * blockDim.x + threadIdx.x;
    if (e < E) atomicAdd(cnt + dst[e], 1.0f);
}

// ---------------------------------------------------------------------------
// Scatter-add: aggr[dst[e], :] += h[src[e], :]   (D = 64, 16 threads/edge x4)
// ---------------------------------------------------------------------------
__global__ void scatter_kernel(const float* __restrict__ h,
                               const int* __restrict__ src,
                               const int* __restrict__ dst,
                               float* __restrict__ aggr, int E) {
    long long gid = (long long)blockIdx.x * blockDim.x + threadIdx.x;
    if (gid >= (long long)E * 16) return;
    int e = (int)(gid >> 4);
    int c = ((int)gid & 15) * 4;
    int s = src[e];
    int d = dst[e];
    float4 v = *(const float4*)(h + (size_t)s * 64 + c);
    float* p = aggr + (size_t)d * 64 + c;
    atomicAdd(p + 0, v.x);
    atomicAdd(p + 1, v.y);
    atomicAdd(p + 2, v.z);
    atomicAdd(p + 3, v.w);
}

// ---------------------------------------------------------------------------
// WMMA GEMM:  C[r, :] = act( [A1[r,:K1] || (A2[r,:K2] * inv(cnt[r]))] @ W + bias )
//   W given transposed bf16: Wt[KOUT][K1+K2]
//   One wave per 16-row tile; 8 waves (256 thr) per block.
//   Fragment layouts per CDNA5 ISA 7.12.2 (wave32).
// ---------------------------------------------------------------------------
template <int K1, int K2, int KOUT, bool RELU, bool MEAN_DIV>
__global__ __launch_bounds__(256)
void gemm_wmma_kernel(const float* __restrict__ A1,
                      const float* __restrict__ A2,
                      const float* __restrict__ cnt,
                      const bf16_t* __restrict__ Wt,
                      const float* __restrict__ bias,
                      float* __restrict__ C, int nrows) {
    constexpr int K   = K1 + K2;
    constexpr int NKT = K / 32;       // k-tiles of 32
    constexpr int NNT = KOUT / 16;    // n-tiles of 16

    const int lane = threadIdx.x & 31;
    const int wave = threadIdx.x >> 5;
    const int tile = blockIdx.x * 8 + wave;
    const int r0   = tile * 16;
    if (r0 >= nrows) return;          // wave-uniform exit (EXEC stays all-1s)

    const int half = lane >> 4;       // 0: K chunk {0..7,16..23}; 1: {8..15,24..31}
    const int mr   = lane & 15;
    int arow = r0 + mr;
    if (arow >= nrows) arow = nrows - 1;   // clamp (N=100000 is 16-aligned anyway)

    float inv = 1.0f;
    if constexpr (MEAN_DIV) {
        inv = 1.0f / fmaxf(cnt[arow], 1.0f);
    }

    // --- Preload all A fragments for this 16-row block (fp32 -> bf16) -------
    v16bf afrag[NKT];
#pragma unroll
    for (int kt = 0; kt < NKT; ++kt) {
        const int kb = kt * 32 + half * 8;
        if constexpr (true) {
            if ((kt + 1) * 32 <= K1) {
                const float4* p0 = (const float4*)(A1 + (size_t)arow * K1 + kb);
                const float4* p1 = (const float4*)(A1 + (size_t)arow * K1 + kb + 16);
                float4 a = p0[0], b = p0[1], c = p1[0], d = p1[1];
                afrag[kt][0]  = (bf16_t)a.x; afrag[kt][1]  = (bf16_t)a.y;
                afrag[kt][2]  = (bf16_t)a.z; afrag[kt][3]  = (bf16_t)a.w;
                afrag[kt][4]  = (bf16_t)b.x; afrag[kt][5]  = (bf16_t)b.y;
                afrag[kt][6]  = (bf16_t)b.z; afrag[kt][7]  = (bf16_t)b.w;
                afrag[kt][8]  = (bf16_t)c.x; afrag[kt][9]  = (bf16_t)c.y;
                afrag[kt][10] = (bf16_t)c.z; afrag[kt][11] = (bf16_t)c.w;
                afrag[kt][12] = (bf16_t)d.x; afrag[kt][13] = (bf16_t)d.y;
                afrag[kt][14] = (bf16_t)d.z; afrag[kt][15] = (bf16_t)d.w;
            } else {
                const int kb2 = kb - K1;   // column within A2
                const float4* p0 = (const float4*)(A2 + (size_t)arow * K2 + kb2);
                const float4* p1 = (const float4*)(A2 + (size_t)arow * K2 + kb2 + 16);
                float4 a = p0[0], b = p0[1], c = p1[0], d = p1[1];
                afrag[kt][0]  = (bf16_t)(a.x * inv); afrag[kt][1]  = (bf16_t)(a.y * inv);
                afrag[kt][2]  = (bf16_t)(a.z * inv); afrag[kt][3]  = (bf16_t)(a.w * inv);
                afrag[kt][4]  = (bf16_t)(b.x * inv); afrag[kt][5]  = (bf16_t)(b.y * inv);
                afrag[kt][6]  = (bf16_t)(b.z * inv); afrag[kt][7]  = (bf16_t)(b.w * inv);
                afrag[kt][8]  = (bf16_t)(c.x * inv); afrag[kt][9]  = (bf16_t)(c.y * inv);
                afrag[kt][10] = (bf16_t)(c.z * inv); afrag[kt][11] = (bf16_t)(c.w * inv);
                afrag[kt][12] = (bf16_t)(d.x * inv); afrag[kt][13] = (bf16_t)(d.y * inv);
                afrag[kt][14] = (bf16_t)(d.z * inv); afrag[kt][15] = (bf16_t)(d.w * inv);
            }
        }
    }

    // --- Sweep output tiles -------------------------------------------------
#pragma unroll
    for (int nt = 0; nt < NNT; ++nt) {
        const int n = nt * 16 + mr;   // this lane's output column (B/C layout)
        const float bv = bias[n];
        v8f c;
#pragma unroll
        for (int i = 0; i < 8; ++i) c[i] = bv;

#pragma unroll
        for (int kt = 0; kt < NKT; ++kt) {
            const int kb = kt * 32 + half * 8;
            const bf16_t* wp = Wt + (size_t)n * K + kb;
            v8bf b0 = *(const v8bf*)(wp);
            v8bf b1 = *(const v8bf*)(wp + 16);
            v16bf bfrag;
#pragma unroll
            for (int j = 0; j < 8; ++j) { bfrag[j] = b0[j]; bfrag[8 + j] = b1[j]; }

            // D = A x B + C  (f32 accumulate)
            c = __builtin_amdgcn_wmma_f32_16x16x32_bf16(
                    /*neg_a=*/false, afrag[kt],
                    /*neg_b=*/false, bfrag,
                    /*c_mod=*/(short)0, c,
                    /*reuse_a=*/false, /*reuse_b=*/false);
        }

        // C layout: lane -> column n; VGPR i -> row r0 + half*8 + i
#pragma unroll
        for (int i = 0; i < 8; ++i) {
            const int row = r0 + half * 8 + i;
            float v = c[i];
            if (RELU) v = fmaxf(v, 0.0f);
            if (row < nrows) C[(size_t)row * KOUT + n] = v;
        }
    }
}

// ---------------------------------------------------------------------------
// Head tail: out[r] = x3[r,:32] . Wl2 + bl2
// ---------------------------------------------------------------------------
__global__ void head_kernel(const float* __restrict__ x3,
                            const float* __restrict__ Wl2,
                            const float* __restrict__ bl2,
                            float* __restrict__ out, int n) {
    int r = blockIdx.x * blockDim.x + threadIdx.x;
    if (r >= n) return;
    const float* row = x3 + (size_t)r * 32;
    float acc = bl2[0];
#pragma unroll
    for (int i = 0; i < 32; ++i) acc += row[i] * Wl2[i];
    out[r] = acc;
}

// ---------------------------------------------------------------------------
// Launch
// ---------------------------------------------------------------------------
extern "C" void kernel_launch(void* const* d_in, const int* in_sizes, int n_in,
                              void* d_out, int out_size, void* d_ws, size_t ws_size,
                              hipStream_t stream) {
    const float* x     = (const float*)d_in[0];
    const int*   eidx  = (const int*)d_in[1];
    // d_in[2] = edge_attr (unused by reference)
    const float* W1a = (const float*)d_in[3];  const float* b1a = (const float*)d_in[4];
    const float* W1b = (const float*)d_in[5];  const float* b1b = (const float*)d_in[6];
    const float* W2a = (const float*)d_in[7];  const float* b2a = (const float*)d_in[8];
    const float* W2b = (const float*)d_in[9];  const float* b2b = (const float*)d_in[10];
    const float* Wl1 = (const float*)d_in[11]; const float* bl1 = (const float*)d_in[12];
    const float* Wl2 = (const float*)d_in[13]; const float* bl2 = (const float*)d_in[14];
    float* out = (float*)d_out;

    const int N = in_sizes[0] / NODES_DIV;     // 100000
    const int E = in_sizes[1] / 2;             // 3.2M
    const int* src = eidx;
    const int* dst = eidx + E;

    // -------- workspace carve-up (all offsets 256B aligned) --------
    char* ws = (char*)d_ws;
    size_t off = 0;
    auto carve = [&](size_t bytes) {
        void* p = ws + off;
        off += (bytes + 255) & ~(size_t)255;
        return p;
    };
    float*  h    = (float*)carve((size_t)N * 64 * 4);   // message buffer (reused both layers)
    float*  aggr = (float*)carve((size_t)N * 64 * 4);   // scatter accumulator (reused)
    float*  cnt  = (float*)carve((size_t)N * 4);
    float*  x1   = (float*)carve((size_t)N * 128 * 4);
    float*  x2   = (float*)carve((size_t)N * 64 * 4);
    float*  x3   = (float*)carve((size_t)N * 32 * 4);
    bf16_t* Wt1a = (bf16_t*)carve((size_t)256 * 64 * 2);
    bf16_t* Wt1b = (bf16_t*)carve((size_t)320 * 128 * 2);
    bf16_t* Wt2a = (bf16_t*)carve((size_t)128 * 64 * 2);
    bf16_t* Wt2b = (bf16_t*)carve((size_t)192 * 64 * 2);
    bf16_t* Wtl1 = (bf16_t*)carve((size_t)64 * 32 * 2);

    const int TB = 256;
    auto cdiv = [](long long a, long long b) { return (int)((a + b - 1) / b); };

    // -------- weight conversion (tiny) --------
    wcvt_kernel<<<cdiv(256 * 64, TB), TB, 0, stream>>>(W1a, Wt1a, 256, 64);
    wcvt_kernel<<<cdiv(320 * 128, TB), TB, 0, stream>>>(W1b, Wt1b, 320, 128);
    wcvt_kernel<<<cdiv(128 * 64, TB), TB, 0, stream>>>(W2a, Wt2a, 128, 64);
    wcvt_kernel<<<cdiv(192 * 64, TB), TB, 0, stream>>>(W2b, Wt2b, 192, 64);
    wcvt_kernel<<<cdiv(64 * 32, TB), TB, 0, stream>>>(Wl1, Wtl1, 64, 32);

    // -------- degree counts (same dst both layers) --------
    hipMemsetAsync(cnt, 0, (size_t)N * 4, stream);
    count_kernel<<<cdiv(E, TB), TB, 0, stream>>>(dst, cnt, E);

    const int gemm_blocks = cdiv((long long)cdiv(N, 16), 8);

    // -------- conv1 --------
    gemm_wmma_kernel<256, 0, 64, false, false><<<gemm_blocks, TB, 0, stream>>>(
        x, nullptr, nullptr, Wt1a, b1a, h, N);
    hipMemsetAsync(aggr, 0, (size_t)N * 64 * 4, stream);
    scatter_kernel<<<cdiv((long long)E * 16, TB), TB, 0, stream>>>(h, src, dst, aggr, E);
    gemm_wmma_kernel<256, 64, 128, true, true><<<gemm_blocks, TB, 0, stream>>>(
        x, aggr, cnt, Wt1b, b1b, x1, N);

    // -------- conv2 --------
    gemm_wmma_kernel<128, 0, 64, false, false><<<gemm_blocks, TB, 0, stream>>>(
        x1, nullptr, nullptr, Wt2a, b2a, h, N);
    hipMemsetAsync(aggr, 0, (size_t)N * 64 * 4, stream);
    scatter_kernel<<<cdiv((long long)E * 16, TB), TB, 0, stream>>>(h, src, dst, aggr, E);
    gemm_wmma_kernel<128, 64, 64, true, true><<<gemm_blocks, TB, 0, stream>>>(
        x1, aggr, cnt, Wt2b, b2b, x2, N);

    // -------- head --------
    gemm_wmma_kernel<64, 0, 32, true, false><<<gemm_blocks, TB, 0, stream>>>(
        x2, nullptr, nullptr, Wtl1, bl1, x3, N);
    head_kernel<<<cdiv(N, TB), TB, 0, stream>>>(x3, Wl2, bl2, out, N);
}
